// Head_16904991277416
// MI455X (gfx1250) — matrix operand order repaired and checked by reading, not compiled, not run on hardware
//
#include <hip/hip_runtime.h>
#include <hip/hip_bf16.h>

typedef __attribute__((ext_vector_type(16))) __bf16 v16bf;
typedef __attribute__((ext_vector_type(8)))  __bf16 v8bf;
typedef __attribute__((ext_vector_type(8)))  float  v8f;
typedef __attribute__((ext_vector_type(4)))  float  v4f;

#define WMMA_BF16(a, b, c) \
    __builtin_amdgcn_wmma_f32_16x16x32_bf16(false, (a), false, (b), (short)0, (c), false, false)

static __device__ __forceinline__ v16bf cat8(v8bf lo, v8bf hi) {
    return __builtin_shufflevector(lo, hi, 0,1,2,3,4,5,6,7,8,9,10,11,12,13,14,15);
}

constexpr int Bn = 4, T = 4096, Cc = 1024, H = 64;
constexpr int BT = Bn * T;                 // 16384 rows
constexpr float NEG = -1e30f;

// ---------------------------------------------------------------------------
// Kernel 0: W[3][C][H] fp32 -> Wt[3][H][C] bf16 (transposed, GEMM-B friendly)
// ---------------------------------------------------------------------------
__global__ void __launch_bounds__(256) wprep(const float* __restrict__ Wq,
                                             const float* __restrict__ Wk,
                                             const float* __restrict__ Wv,
                                             __bf16* __restrict__ Wt) {
    int idx = blockIdx.x * 256 + threadIdx.x;          // [w][h][c] flat
    if (idx >= 3 * H * Cc) return;
    int w = idx >> 16;                                 // H*C = 65536
    int h = (idx >> 10) & (H - 1);
    int c = idx & (Cc - 1);
    const float* W = (w == 0) ? Wq : ((w == 1) ? Wk : Wv);
    Wt[idx] = (__bf16)W[c * H + h];
}

// ---------------------------------------------------------------------------
// Kernel 1: q/k/v projections. One wave = one 16x64 output tile.
//   Q,K stored [BT][64] bf16 ; V stored transposed [b][64][T] bf16.
// ---------------------------------------------------------------------------
__global__ void __launch_bounds__(256) qkv(const float*  __restrict__ x,
                                           const __bf16* __restrict__ Wt,
                                           __bf16* __restrict__ Qb,
                                           __bf16* __restrict__ Kb,
                                           __bf16* __restrict__ Vt) {
    const int lane = threadIdx.x & 31;
    const int wid  = blockIdx.x * 8 + (threadIdx.x >> 5);
    const int w    = wid >> 10;            // 0=q 1=k 2=v
    const int tile = wid & 1023;           // 16-row tile of BT
    const int row0 = tile * 16;
    const int mrow = lane & 15;
    const int half = lane >> 4;

    const v8f z8 = {0.f,0.f,0.f,0.f,0.f,0.f,0.f,0.f};
    v8f acc[4] = {z8, z8, z8, z8};

    const float*  xrow = x  + (size_t)(row0 + mrow) * Cc;
    const __bf16* wb   = Wt + (size_t)w * H * Cc;

    for (int kb = 0; kb < Cc; kb += 32) {
        // A fragment: 16x32 bf16 from x (fp32 -> bf16 convert on the fly)
        const float* xp = xrow + kb + half * 8;
        v4f f0 = *(const v4f*)(xp);
        v4f f1 = *(const v4f*)(xp + 4);
        v4f f2 = *(const v4f*)(xp + 16);
        v4f f3 = *(const v4f*)(xp + 20);
        v16bf a;
#pragma unroll
        for (int i = 0; i < 4; ++i) {
            a[i]      = (__bf16)f0[i];
            a[4 + i]  = (__bf16)f1[i];
            a[8 + i]  = (__bf16)f2[i];
            a[12 + i] = (__bf16)f3[i];
        }
#pragma unroll
        for (int nt = 0; nt < 4; ++nt) {
            // B fragment: Wt[w][nt*16+lane][kb + half*8 .. +7] and [kb + 16 + half*8 .. +7]
            const __bf16* bp = wb + (size_t)(nt * 16 + mrow) * Cc + kb + half * 8;
            v16bf bf = cat8(*(const v8bf*)bp, *(const v8bf*)(bp + 16));
            acc[nt] = WMMA_BF16(a, bf, acc[nt]);
        }
    }

    if (w < 2) {
        __bf16* dst = (w == 0) ? Qb : Kb;
#pragma unroll
        for (int nt = 0; nt < 4; ++nt)
#pragma unroll
            for (int j = 0; j < 8; ++j) {
                int r = row0 + half * 8 + j;
                dst[(size_t)r * H + nt * 16 + mrow] = (__bf16)acc[nt][j];
            }
    } else {
#pragma unroll
        for (int nt = 0; nt < 4; ++nt)
#pragma unroll
            for (int j = 0; j < 8; ++j) {
                int r  = row0 + half * 8 + j;
                int b  = r >> 12;          // /4096
                int t  = r & (T - 1);
                Vt[(size_t)b * H * T + (size_t)(nt * 16 + mrow) * T + t] = (__bf16)acc[nt][j];
            }
    }
}

// ---------------------------------------------------------------------------
// Kernel 2: causal flash attention. One wave = 16 query rows; k-blocks of 32.
// ---------------------------------------------------------------------------
__global__ void __launch_bounds__(128) attn(const __bf16* __restrict__ Qb,
                                            const __bf16* __restrict__ Kb,
                                            const __bf16* __restrict__ Vt,
                                            float* __restrict__ out) {
    __shared__ __attribute__((aligned(16))) __bf16 plds[4][16 * 32];

    const int lane = threadIdx.x & 31;
    const int widx = threadIdx.x >> 5;
    const int qt   = blockIdx.x * 4 + widx;      // 0..1023
    const int b    = qt >> 8;                    // 256 q-tiles per batch
    const int qb   = (qt & 255) * 16;            // first query row (within batch)
    const int mrow = lane & 15;
    const int half = lane >> 4;

    // Q fragments for H-chunks {0..31},{32..63}; reused all k-blocks
    v16bf qf[2];
    {
        const __bf16* qp = Qb + (size_t)(b * T + qb + mrow) * H;
#pragma unroll
        for (int h0 = 0; h0 < 2; ++h0) {
            v8bf lo = *(const v8bf*)(qp + h0 * 32 + half * 16);
            v8bf hi = *(const v8bf*)(qp + h0 * 32 + half * 16 + 8);
            qf[h0] = cat8(lo, hi);
        }
    }

    const v8f z8 = {0.f,0.f,0.f,0.f,0.f,0.f,0.f,0.f};
    v8f O[4] = {z8, z8, z8, z8};
    float m[8], l[8];
#pragma unroll
    for (int j = 0; j < 8; ++j) { m[j] = NEG; l[j] = 0.f; }

    const __bf16* kbat = Kb + (size_t)b * T * H;
    const __bf16* vbat = Vt + (size_t)b * H * T;
    const int nblocks = (qb + 15) / 32 + 1;

    for (int jb = 0; jb < nblocks; ++jb) {
        const int k0 = jb * 32;

        // ---- S = Q @ K^T (two 16x16 column tiles of a 16x32 S strip) ----
        v8f S0 = z8, S1 = z8;
#pragma unroll
        for (int h0 = 0; h0 < 2; ++h0) {
            const __bf16* kp0 = kbat + (size_t)(k0 + mrow) * H + h0 * 32 + half * 16;
            const __bf16* kp1 = kbat + (size_t)(k0 + 16 + mrow) * H + h0 * 32 + half * 16;
            v16bf bk0 = cat8(*(const v8bf*)kp0, *(const v8bf*)(kp0 + 8));
            v16bf bk1 = cat8(*(const v8bf*)kp1, *(const v8bf*)(kp1 + 8));
            S0 = WMMA_BF16(qf[h0], bk0, S0);
            S1 = WMMA_BF16(qf[h0], bk1, S1);
        }

        // ---- scale, causal mask, online softmax ----
        const bool needmask = (k0 + 31 > qb);
        float p0[8], p1[8], cf[8];
#pragma unroll
        for (int j = 0; j < 8; ++j) {
            float s0 = S0[j] * 0.125f;               // 1/sqrt(64)
            float s1 = S1[j] * 0.125f;
            if (needmask) {
                int r = qb + half * 8 + j;           // query index
                if (k0 + mrow > r)      s0 = NEG;
                if (k0 + 16 + mrow > r) s1 = NEG;
            }
            float rx = fmaxf(s0, s1);                // row max across 16 lanes of half
#pragma unroll
            for (int off = 1; off < 16; off <<= 1)
                rx = fmaxf(rx, __shfl_xor(rx, off, 32));
            float mn = fmaxf(m[j], rx);
            float c  = __expf(m[j] - mn);
            float e0 = __expf(s0 - mn);
            float e1 = __expf(s1 - mn);
            float rs = e0 + e1;
#pragma unroll
            for (int off = 1; off < 16; off <<= 1)
                rs += __shfl_xor(rs, off, 32);
            l[j] = l[j] * c + rs;
            m[j] = mn;
            cf[j] = c;
            p0[j] = e0;
            p1[j] = e1;
        }
#pragma unroll
        for (int nt = 0; nt < 4; ++nt)
#pragma unroll
            for (int j = 0; j < 8; ++j) O[nt][j] *= cf[j];

        // ---- P: C-layout f32 -> A-fragment bf16 via per-wave LDS tile ----
#pragma unroll
        for (int j = 0; j < 8; ++j) {
            int r = half * 8 + j;
            plds[widx][r * 32 + mrow]      = (__bf16)p0[j];
            plds[widx][r * 32 + 16 + mrow] = (__bf16)p1[j];
        }
        asm volatile("s_wait_dscnt 0" ::: "memory");   // same-wave LDS RAW
        v16bf pf;
        {
            const __bf16* pp = &plds[widx][mrow * 32 + half * 16];
            pf = cat8(*(const v8bf*)pp, *(const v8bf*)(pp + 8));
        }

        // ---- O += P @ V  (V stored transposed: contiguous over keys) ----
#pragma unroll
        for (int nt = 0; nt < 4; ++nt) {
            const __bf16* vp = vbat + (size_t)(nt * 16 + mrow) * T + k0 + half * 16;
            v16bf bv = cat8(*(const v8bf*)vp, *(const v8bf*)(vp + 8));
            O[nt] = WMMA_BF16(pf, bv, O[nt]);
        }
    }

    // ---- epilogue: divide by row sums, store fp32 ----
#pragma unroll
    for (int nt = 0; nt < 4; ++nt)
#pragma unroll
        for (int j = 0; j < 8; ++j) {
            int r = qb + half * 8 + j;
            out[(size_t)(b * T + r) * H + nt * 16 + mrow] = O[nt][j] / l[j];
        }
}

// ---------------------------------------------------------------------------
extern "C" void kernel_launch(void* const* d_in, const int* in_sizes, int n_in,
                              void* d_out, int out_size, void* d_ws, size_t ws_size,
                              hipStream_t stream) {
    const float* x  = (const float*)d_in[0];
    const float* Wq = (const float*)d_in[1];
    const float* Wk = (const float*)d_in[2];
    const float* Wv = (const float*)d_in[3];
    float* out = (float*)d_out;

    char* ws = (char*)d_ws;
    const size_t WT_SZ = (size_t)3 * H * Cc * sizeof(__bf16);   // 384 KB
    const size_t QK_SZ = (size_t)BT * H * sizeof(__bf16);       // 2 MB each
    __bf16* Wt = (__bf16*)ws;
    __bf16* Qb = (__bf16*)(ws + WT_SZ);
    __bf16* Kb = (__bf16*)(ws + WT_SZ + QK_SZ);
    __bf16* Vt = (__bf16*)(ws + WT_SZ + 2 * QK_SZ);

    wprep<<<(3 * H * Cc + 255) / 256, 256, 0, stream>>>(Wq, Wk, Wv, Wt);
    qkv<<<(3 * (BT / 16)) / 8, 256, 0, stream>>>(x, Wt, Qb, Kb, Vt);
    attn<<<(BT / 16) / 4, 128, 0, stream>>>(Qb, Kb, Vt, out);
}
